// memorySlots_64424509440862
// MI455X (gfx1250) — matrix operand order repaired and checked by reading, not compiled, not run on hardware
//
#include <hip/hip_runtime.h>
#include <hip/hip_bf16.h>

typedef __attribute__((ext_vector_type(16))) __bf16 v16bf;
typedef __attribute__((ext_vector_type(8)))  __bf16 v8bf;
typedef __attribute__((ext_vector_type(8)))  float  v8f;

#define BN_EPS_F 1e-5f

// round-to-nearest-even f32 -> bf16 (stored as ushort)
static __device__ inline unsigned short f2bf(float f) {
    unsigned u = __float_as_uint(f);
    unsigned r = (u + 0x7FFFu + ((u >> 16) & 1u)) >> 16;
    return (unsigned short)r;
}

// 32B fragment load as two 16B halves (LDS pitch only guarantees 16B alignment)
static __device__ inline v16bf load16bf(const unsigned short* p) {
    union { v16bf v; v8bf h[2]; } u;
    u.h[0] = *(const v8bf*)(p);
    u.h[1] = *(const v8bf*)(p + 8);
    return u.v;
}

// ---------------- prep 1: h2 = relu(relu(w1 + b1) @ w2 + b2), 13x2048 ----------------
__global__ void prep_h2_kernel(const float* __restrict__ w1, const float* __restrict__ b1,
                               const float* __restrict__ w2, const float* __restrict__ b2,
                               float* __restrict__ h2) {
    int idx = blockIdx.x * 256 + threadIdx.x;
    if (idx >= 13 * 2048) return;
    int s = idx >> 11, q = idx & 2047;
    float acc = b2[q];
    for (int p = 0; p < 1024; ++p) {
        float h = w1[s * 1024 + p] + b1[p];
        h = h > 0.f ? h : 0.f;
        acc = fmaf(h, w2[p * 2048 + q], acc);
    }
    h2[idx] = acc > 0.f ? acc : 0.f;
}

// ---------------- prep 2: f, normalize, g = f@conv2_w, bs = f@conv2_b, ----------------
// ---------------- fold BN into conv1 weights and pre-swizzle WMMA A-fragments --------
__global__ void prep_rest_kernel(const float* __restrict__ h2, const float* __restrict__ w3,
                                 const float* __restrict__ b3,
                                 const float* __restrict__ conv1_w, const float* __restrict__ conv1_b,
                                 const float* __restrict__ gamma, const float* __restrict__ beta,
                                 const float* __restrict__ mean, const float* __restrict__ var,
                                 const float* __restrict__ conv2_w, const float* __restrict__ conv2_b,
                                 float* __restrict__ out_f, unsigned short* __restrict__ wA,
                                 unsigned short* __restrict__ gA, float* __restrict__ b1f,
                                 float* __restrict__ bs) {
    __shared__ float f_lds[13 * 96];
    __shared__ float g_lds[13 * 32];
    __shared__ float nrm[13];
    const int tid = threadIdx.x;

    for (int idx = tid; idx < 13 * 96; idx += 256) {
        int s = idx / 96, c = idx % 96;
        float acc = b3[c];
        for (int q = 0; q < 2048; ++q) acc = fmaf(h2[s * 2048 + q], w3[q * 96 + c], acc);
        f_lds[idx] = acc;
    }
    __syncthreads();
    if (tid < 13) {
        float nn = 0.f;
        for (int c = 0; c < 96; ++c) { float v = f_lds[tid * 96 + c]; nn += v * v; }
        nrm[tid] = fmaxf(sqrtf(nn), 1e-12f);
    }
    __syncthreads();
    for (int idx = tid; idx < 13 * 96; idx += 256) {
        float v = f_lds[idx] / nrm[idx / 96];
        f_lds[idx] = v;
        out_f[idx] = v;   // memory_filters output (1,13,96)
    }
    __syncthreads();
    for (int idx = tid; idx < 13 * 32; idx += 256) {
        int s = idx / 32, ci = idx % 32;
        float acc = 0.f;
        for (int co = 0; co < 96; ++co) acc = fmaf(f_lds[s * 96 + co], conv2_w[co * 32 + ci], acc);
        g_lds[idx] = acc;
    }
    if (tid < 13) {
        float acc = 0.f;
        for (int co = 0; co < 96; ++co) acc = fmaf(f_lds[tid * 96 + co], conv2_b[co], acc);
        bs[tid] = acc;
    }
    if (tid < 32) {
        float inv = gamma[tid] * rsqrtf(var[tid] + BN_EPS_F);
        b1f[tid] = (conv1_b[tid] - mean[tid]) * inv + beta[tid];
    }
    __syncthreads();
    // gA: 16x32 bf16 A-fragment layout (M=slot rows, rows 13..15 zero)
    // lane L: m=L&15, khalf=L>>4 ; element e -> K = e + 16*khalf
    for (int idx = tid; idx < 512; idx += 256) {
        int lane = idx >> 4, e = idx & 15;
        int kh = lane >> 4, m = lane & 15;
        int cin = e + 16 * kh;
        float v = (m < 13) ? g_lds[m * 32 + cin] : 0.f;
        gA[idx] = f2bf(v);
    }
    // wA[mhalf][tap][lane][e], BN scale folded into weights; tap = (kz*3+ky)*3+kx
    for (int idx = tid; idx < 2 * 27 * 32 * 16; idx += 256) {
        int e = idx & 15;
        int lane = (idx >> 4) & 31;
        int tap = (idx >> 9) % 27;
        int mh = idx / (512 * 27);
        int kh = lane >> 4, m = lane & 15;
        int cout = mh * 16 + m;
        int cin = e + 16 * kh;
        float inv = gamma[cout] * rsqrtf(var[cout] + BN_EPS_F);
        wA[idx] = f2bf(conv1_w[(cout * 32 + cin) * 27 + tap] * inv);
    }
}

// ---------------- main fused kernel: conv1(+BN+PReLU) -> logits -----------------------
// block covers 2 z-planes x one y-row x 96 x.  384 threads = 12 waves:
//   wave w -> zsub = w/6, mhalf = (w%6)/3, npair = w%3 ; each wave owns 2 x-subtiles.
__global__ __launch_bounds__(384) void conv_logits_kernel(
    const float* __restrict__ y, const unsigned short* __restrict__ wA,
    const unsigned short* __restrict__ gA, const float* __restrict__ b1f,
    const float* __restrict__ bs, const float* __restrict__ alpha_p,
    float* __restrict__ out_logits) {
    // input row staging, double buffered: [buf][x 0..97][cin], bf16, pitch 40
    __shared__ unsigned short lds_in[2][98][40];
    // t tiles after bias+prelu: [zsub][x][cout], bf16
    __shared__ unsigned short lds_t[2][96][40];

    const int z0 = blockIdx.x * 2, yrow0 = blockIdx.y, b = blockIdx.z;
    const int tid = threadIdx.x;
    const int lane = tid & 31, w = tid >> 5;
    const int zsub = w / 6;
    const int mh = (w % 6) / 3;
    const int npair = w % 3;
    const int n = lane & 15, kh = lane >> 4;
    const int x0 = npair * 32;            // wave's two subtiles: x0+{0..15}, x0+{16..31}

    // 12 input rows: ri -> (zi = ri/3 in 0..3 -> zz = z0+zi-1, ky = ri%3 -> yy = yrow0+ky-1)
    auto stage = [&](int ri, int buf) {
        int zi = ri / 3, ky = ri % 3;
        int zz = z0 + zi - 1, yy = yrow0 + ky - 1;
        bool rowok = ((unsigned)zz < 96u) && ((unsigned)yy < 96u);
        const float* src = y + ((((size_t)b * 32) * 96 + (size_t)(rowok ? zz : 0)) * 96 +
                                (size_t)(rowok ? yy : 0)) * 96;
        for (int idx = tid; idx < 98 * 32; idx += 384) {
            int cin = idx / 98, x = idx - cin * 98;
            int xx = x - 1;
            float v = 0.f;
            if (rowok && (unsigned)xx < 96u)
                v = src[(size_t)cin * (96 * 96 * 96) + xx];
            lds_in[buf][x][cin] = f2bf(v);
        }
    };

    v8f acc0 = {};
    v8f acc1 = {};
    stage(0, 0);
    for (int ri = 0; ri < 12; ++ri) {
        __syncthreads();                         // stage(ri) visible
        if (ri < 11) stage(ri + 1, (ri + 1) & 1);
        const int buf = ri & 1;
        const int zi = ri / 3, ky = ri % 3;
        const int kz = zi - zsub;                // wave-uniform
        if ((unsigned)kz < 3u) {
#pragma unroll
            for (int kx = 0; kx < 3; ++kx) {
                const int tap = (kz * 3 + ky) * 3 + kx;
                v16bf a  = load16bf(wA + ((size_t)(mh * 27 + tap) * 32 + lane) * 16);
                v16bf b0 = load16bf(&lds_in[buf][x0 + n + kx][kh * 16]);
                v16bf b1 = load16bf(&lds_in[buf][x0 + 16 + n + kx][kh * 16]);
                acc0 = __builtin_amdgcn_wmma_f32_16x16x32_bf16(
                    false, a, false, b0, (short)0, acc0, false, false);
                acc1 = __builtin_amdgcn_wmma_f32_16x16x32_bf16(
                    false, a, false, b1, (short)0, acc1, false, false);
            }
        }
    }

    // epilogue: fused BN bias + PReLU, spill both t subtiles (bf16) for second GEMM
    const float alpha = alpha_p[0];
    const int coutbase = mh * 16 + 8 * kh;  // cout = coutbase + j (D-layout: M = j + 8*khalf)
#pragma unroll
    for (int j = 0; j < 8; ++j) {
        float bb = b1f[coutbase + j];
        float v0 = acc0[j] + bb;
        float v1 = acc1[j] + bb;
        v0 = v0 >= 0.f ? v0 : alpha * v0;
        v1 = v1 >= 0.f ? v1 : alpha * v1;
        lds_t[zsub][x0 + n][coutbase + j]      = f2bf(v0);
        lds_t[zsub][x0 + 16 + n][coutbase + j] = f2bf(v1);
    }
    __syncthreads();

    // logits[s, x] = g @ t + bs : one WMMA per 16-column subtile, all 12 waves
    {
        const int zsub2 = w / 6, nsub2 = w % 6;
        v16bf ga = load16bf(gA + (size_t)lane * 16);
        v16bf bm = load16bf(&lds_t[zsub2][nsub2 * 16 + n][kh * 16]);
        v8f acc2 = {};
        acc2 = __builtin_amdgcn_wmma_f32_16x16x32_bf16(
            false, ga, false, bm, (short)0, acc2, false, false);
        const int x = nsub2 * 16 + n;
        const int z = z0 + zsub2;
#pragma unroll
        for (int j = 0; j < 8; ++j) {
            int s = j + 8 * kh;
            if (s < 13) {
                size_t off = ((((size_t)b * 13 + s) * 96 + z) * 96 + yrow0) * 96 + x;
                out_logits[off] = acc2[j] + bs[s];
            }
        }
    }
}

extern "C" void kernel_launch(void* const* d_in, const int* in_sizes, int n_in,
                              void* d_out, int out_size, void* d_ws, size_t ws_size,
                              hipStream_t stream) {
    const float* y       = (const float*)d_in[0];
    const float* conv1_w = (const float*)d_in[1];
    const float* conv1_b = (const float*)d_in[2];
    const float* gamma   = (const float*)d_in[3];
    const float* beta    = (const float*)d_in[4];
    const float* mean    = (const float*)d_in[5];
    const float* var     = (const float*)d_in[6];
    const float* alpha   = (const float*)d_in[7];
    const float* conv2_w = (const float*)d_in[8];
    const float* conv2_b = (const float*)d_in[9];
    const float* w1 = (const float*)d_in[10];
    const float* b1 = (const float*)d_in[11];
    const float* w2 = (const float*)d_in[12];
    const float* b2 = (const float*)d_in[13];
    const float* w3 = (const float*)d_in[14];
    const float* b3 = (const float*)d_in[15];

    // workspace layout (all offsets 32B-aligned)
    char* ws = (char*)d_ws;
    float*          h2  = (float*)(ws + 0);                          // 13*2048*4 = 106496
    unsigned short* wA  = (unsigned short*)(ws + 106496);            // 2*27*32*16*2 = 55296
    unsigned short* gA  = (unsigned short*)(ws + 106496 + 55296);    // 512*2 = 1024
    float*          b1f = (float*)(ws + 106496 + 55296 + 1024);      // 32*4
    float*          bsv = (float*)(ws + 106496 + 55296 + 1024 + 128);// 13*4

    float* out_f      = (float*)d_out;        // memory_filters: 13*96
    float* out_logits = out_f + 13 * 96;      // logits: 2*13*96^3

    prep_h2_kernel<<<dim3((13 * 2048 + 255) / 256), 256, 0, stream>>>(w1, b1, w2, b2, h2);
    prep_rest_kernel<<<1, 256, 0, stream>>>(h2, w3, b3, conv1_w, conv1_b, gamma, beta,
                                            mean, var, conv2_w, conv2_b,
                                            out_f, wA, gA, b1f, bsv);
    conv_logits_kernel<<<dim3(48, 96, 2), 384, 0, stream>>>(y, wA, gA, b1f, bsv, alpha,
                                                            out_logits);
}
